// Encoder_81080392614207
// MI455X (gfx1250) — matrix operand order repaired and checked by reading, not compiled, not run on hardware
//
#include <hip/hip_runtime.h>
#include <hip/hip_bf16.h>

// ---------------------------------------------------------------------------
// CSI encoder for MI455X (gfx1250, wave32, WMMA).
//
// Single unified GEMM:
//   A [368 x 768]  : selected/reshaped x rows (23 tiles of 16), zeros at
//                    inactive j positions for the j<4 / j>=4 groups.
//   B [768 x 32768]: codebook entries (mirror entries handled analytically).
//   score_lo(k)  = csq_g[k]                - 2*cross
//   score_hi(k)  = (Ng - 2*csum_g + csq_g) - 2*rowsumA[m] + 2*cross
// argmin over 65536 candidates per output row, then 22-bit expansion.
//
// Precision: bf16 hi/lo split, 3x v_wmma_f32_16x16x32_bf16 per K-step.
// Loop nest: K-step outer, 23 M-tiles inner with 23 resident v8f
// accumulators; B fragment converted once per K-step and reused across
// 69 WMMAs; A stream double-buffered.  No LDS in the GEMM kernel.
// ---------------------------------------------------------------------------

typedef __attribute__((ext_vector_type(16))) __bf16       v16bf;
typedef __attribute__((ext_vector_type(8)))  float        v8f;
typedef __attribute__((ext_vector_type(4)))  unsigned int u32x4;

#define NTILES   23
#define KSTEPS   24          // 768 / 32
#define NROWS    368         // 23 * 16
#define NWAVES   2048        // 32768 entries / 16 per wave
#define FRAG_B   64          // bytes per lane per fragment pair (hi16+lo16 bf16)

// workspace layout (bytes)
#define WS_APACK   0
#define APACK_SZ   (KSTEPS * NTILES * 32 * FRAG_B)   // 1,130,496
#define WS_ROWSUM  1130496
#define WS_CANDV   1132032
#define CAND_SZ    (NROWS * NWAVES * 4)              // 3,014,656
#define WS_CANDI   (1132032 + 3014656)

// ---- helpers --------------------------------------------------------------

__device__ __forceinline__ void bf16_split(float c, unsigned& hib, unsigned& lob) {
    hib = __float_as_uint(c) & 0xFFFF0000u;            // truncate-to-bf16 (hi)
    float hf = __uint_as_float(hib);
    lob = __float_as_uint(c - hf) & 0xFFFF0000u;       // residual, truncated
}

__device__ __forceinline__ void lexmin(float& v, int& i, float v2, int i2) {
    if (v2 < v || (v2 == v && i2 < i)) { v = v2; i = i2; }
}

__device__ __forceinline__ void row_map(int m, int& a, int& c) {
    int t = m >> 4, mr = m & 15;
    if (t < 7)       { int r = t * 16 + mr;        a = r / 7; c = r % 7; }
    else if (t < 14) { int r = (t - 7) * 16 + mr;  a = r / 7; c = 7 + r % 7; }
    else             { int r = (t - 14) * 16 + mr; a = r / 9; c = 14 + r % 9; }
}

// ---- phase 1a: pack A into WMMA 16-bit A-operand layout (hi/lo bf16) ------
// Layout: apack[s][t][lane][64B]  (K-step major so the inner t-loop of the
// GEMM walks contiguous 2KB strides).  Lane l: row mr=l&15, half h=l>>4.
// A element e (0..15) -> K = 16h + e  (contiguous; same order as B).

__global__ __launch_bounds__(32) void csi_pack_a(const float* __restrict__ x,
                                                 unsigned char* __restrict__ apack) {
    const int blk = blockIdx.x;              // s*NTILES + t
    const int s = blk / NTILES, t = blk % NTILES;
    const int lane = threadIdx.x;
    const int mr = lane & 15, h = lane >> 4;

    int g, rowid;
    if (t < 7)       { g = 0; rowid = t * 16 + mr; }
    else if (t < 14) { g = 1; rowid = (t - 7) * 16 + mr; }
    else             { g = 2; rowid = (t - 14) * 16 + mr; }
    int a, i;
    if (g < 2) { a = rowid / 7; i = rowid % 7; }
    else       { a = rowid / 9; i = 7 + rowid % 9; }

    unsigned hiu[8], lou[8];
#pragma unroll
    for (int e = 0; e < 16; ++e) {
        int kap = s * 32 + 16 * h + e;
        int p   = kap / 384;
        int rem = kap - p * 384;
        int q   = rem >> 3;
        int j   = kap & 7;
        bool active = (g == 0) ? (j < 4) : ((g == 1) ? (j >= 4) : true);
        float val = 0.0f;
        if (active)
            val = x[(((size_t)a * 2 + p) * 126 + (19 + q)) * 128 + i * 8 + j];
        unsigned hib, lob;
        bf16_split(val, hib, lob);
        if (e & 1) { hiu[e >> 1] |= hib;       lou[e >> 1] |= lob; }
        else       { hiu[e >> 1]  = hib >> 16; lou[e >> 1]  = lob >> 16; }
    }
    u32x4* dst = (u32x4*)(apack + ((size_t)blk * 32 + lane) * FRAG_B);
    u32x4 q0, q1, q2, q3;
    q0.x = hiu[0]; q0.y = hiu[1]; q0.z = hiu[2]; q0.w = hiu[3];
    q1.x = hiu[4]; q1.y = hiu[5]; q1.z = hiu[6]; q1.w = hiu[7];
    q2.x = lou[0]; q2.y = lou[1]; q2.z = lou[2]; q2.w = lou[3];
    q3.x = lou[4]; q3.y = lou[5]; q3.z = lou[6]; q3.w = lou[7];
    dst[0] = q0; dst[1] = q1; dst[2] = q2; dst[3] = q3;
}

// ---- phase 1b: per-row sums of A (active positions only) ------------------

__global__ __launch_bounds__(256) void csi_rowsum(const float* __restrict__ x,
                                                  float* __restrict__ rowsum) {
    int m = blockIdx.x * blockDim.x + threadIdx.x;
    if (m >= NROWS) return;
    int t = m >> 4, mr = m & 15;
    int g, rowid;
    if (t < 7)       { g = 0; rowid = t * 16 + mr; }
    else if (t < 14) { g = 1; rowid = (t - 7) * 16 + mr; }
    else             { g = 2; rowid = (t - 14) * 16 + mr; }
    int a, i;
    if (g < 2) { a = rowid / 7; i = rowid % 7; }
    else       { a = rowid / 9; i = 7 + rowid % 9; }
    int jlo = (g == 1) ? 4 : 0;
    int jhi = (g == 0) ? 4 : 8;
    float sum = 0.0f;
    for (int p = 0; p < 2; ++p)
        for (int q = 0; q < 48; ++q)
            for (int j = jlo; j < jhi; ++j)
                sum += x[(((size_t)a * 2 + p) * 126 + (19 + q)) * 128 + i * 8 + j];
    rowsum[m] = sum;
}

// ---- phase 2: main WMMA GEMM + per-wave argmin ----------------------------
// grid: 512 blocks x 128 threads (4 waves), no LDS.  Each wave owns a
// 16-entry N-tile and keeps 23 f32 accumulator tiles resident.  Per
// K-step it converts its 32 B values to bf16 hi/lo once and reuses them
// for 23 tiles x 3 WMMAs, streaming A fragments with a double buffer.

__global__ __launch_bounds__(128) void csi_gemm(const float* __restrict__ codebook,
                                                const unsigned char* __restrict__ apack,
                                                const float* __restrict__ rowsum,
                                                float* __restrict__ candV,
                                                int* __restrict__ candI) {
    const int tid  = threadIdx.x;
    const int lane = tid & 31;
    const int w    = tid >> 5;
    const int n    = lane & 15;       // N column within tile
    const int h    = lane >> 4;       // lane half
    const int wave_id = blockIdx.x * 4 + w;
    const int k_base  = wave_id * 16; // first codebook entry of this wave

    // B-operand layout: lane = column n, half h covers K = 16h..16h+15,
    // element e -> K = 16h + e  (contiguous 16 floats per K-step per lane).
    const float* ebase = codebook + (size_t)(k_base + n) * 768 + 16 * h;
    const unsigned char* alane = apack + lane * FRAG_B;

    union frag { v16bf v; u32x4 q[2]; };

    float csq1 = 0.f, csq2 = 0.f, csum1 = 0.f, csum2 = 0.f;

    v8f acc[NTILES];
#pragma unroll
    for (int t = 0; t < NTILES; ++t)
#pragma unroll
        for (int r = 0; r < 8; ++r) acc[t][r] = 0.0f;

#pragma unroll 1
    for (int s = 0; s < KSTEPS; ++s) {
        // ---- convert this K-step's B column chunk (32 values: 2 halves) ---
        const float4* src = (const float4*)(ebase + s * 32);
        frag bh, bl;
#pragma unroll
        for (int c4 = 0; c4 < 4; ++c4) {
            float4 f = src[c4];
            float vv[4] = {f.x, f.y, f.z, f.w};
            unsigned pk_h = 0, pk_l = 0;
#pragma unroll
            for (int u = 0; u < 4; ++u) {
                int e = c4 * 4 + u;
                int j = e & 7;                       // (16h) % 8 == 0
                float c = vv[u];
                if (j < 4) { csq1 += c * c; csum1 += c; }
                else       { csq2 += c * c; csum2 += c; }
                unsigned hib, lob;
                bf16_split(c, hib, lob);
                if (u & 1) { pk_h |= hib;       pk_l |= lob;
                             if (u == 3) { bh.q[c4 >> 1][2 * (c4 & 1) + 1] = pk_h;
                                           bl.q[c4 >> 1][2 * (c4 & 1) + 1] = pk_l; }
                             else        { bh.q[c4 >> 1][2 * (c4 & 1)]     = pk_h;
                                           bl.q[c4 >> 1][2 * (c4 & 1)]     = pk_l;
                                           pk_h = 0; pk_l = 0; } }
                else       { pk_h  = hib >> 16; pk_l  = lob >> 16; }
            }
        }

        // ---- 23 M-tiles, A stream double-buffered (distance 2) ------------
        const unsigned char* abase = alane + (size_t)s * (NTILES * 32 * FRAG_B);
        frag ah[2], al[2];
        {
            const u32x4* ap0 = (const u32x4*)(abase);
            ah[0].q[0] = ap0[0]; ah[0].q[1] = ap0[1];
            al[0].q[0] = ap0[2]; al[0].q[1] = ap0[3];
            const u32x4* ap1 = (const u32x4*)(abase + 32 * FRAG_B);
            ah[1].q[0] = ap1[0]; ah[1].q[1] = ap1[1];
            al[1].q[0] = ap1[2]; al[1].q[1] = ap1[3];
        }
#pragma unroll
        for (int t = 0; t < NTILES; ++t) {
            const int buf = t & 1;
            // cross ~= A_hi*B_hi + A_hi*B_lo + A_lo*B_hi  (fp32 accumulate)
            acc[t] = __builtin_amdgcn_wmma_f32_16x16x32_bf16(false, ah[buf].v, false, bh.v,
                                                             (short)0, acc[t], false, false);
            acc[t] = __builtin_amdgcn_wmma_f32_16x16x32_bf16(false, ah[buf].v, false, bl.v,
                                                             (short)0, acc[t], false, false);
            acc[t] = __builtin_amdgcn_wmma_f32_16x16x32_bf16(false, al[buf].v, false, bh.v,
                                                             (short)0, acc[t], false, false);
            if (t + 2 < NTILES) {
                const u32x4* ap = (const u32x4*)(abase + (size_t)(t + 2) * (32 * FRAG_B));
                ah[buf].q[0] = ap[0]; ah[buf].q[1] = ap[1];
                al[buf].q[0] = ap[2]; al[buf].q[1] = ap[3];
            }
        }
    }

    // ---- epilogue: scores + argmin over 16 entries (x2 with mirror) -------
    csq1  += __shfl_xor(csq1, 16);
    csq2  += __shfl_xor(csq2, 16);
    csum1 += __shfl_xor(csum1, 16);
    csum2 += __shfl_xor(csum2, 16);

    const float blo[3] = { csq1, csq2, csq1 + csq2 };
    const float bhi[3] = { 384.f - 2.f * csum1 + csq1,
                           384.f - 2.f * csum2 + csq2,
                           768.f - 2.f * (csum1 + csum2) + csq1 + csq2 };

#pragma unroll 1
    for (int t = 0; t < NTILES; ++t) {
        const int g = (t < 7) ? 0 : ((t < 14) ? 1 : 2);
        const float4* rsp = (const float4*)(rowsum + t * 16 + 8 * h);
        float4 rs0 = rsp[0];
        float4 rs1 = rsp[1];
        float rs8[8] = { rs0.x, rs0.y, rs0.z, rs0.w, rs1.x, rs1.y, rs1.z, rs1.w };
#pragma unroll
        for (int r = 0; r < 8; ++r) {
            float cross = acc[t][r];
            float rs = rs8[r];
            float v  = blo[g] - 2.f * cross;                 int idx  = k_base + n;
            float v2 = bhi[g] - 2.f * rs + 2.f * cross;      int idx2 = idx + 32768;
            lexmin(v, idx, v2, idx2);
#pragma unroll
            for (int m = 1; m < 16; m <<= 1) {
                float vv = __shfl_xor(v, m);
                int   ii = __shfl_xor(idx, m);
                lexmin(v, idx, vv, ii);
            }
            if (n == 0) {
                int row = t * 16 + 8 * h + r;
                candV[(size_t)row * NWAVES + wave_id] = v;
                candI[(size_t)row * NWAVES + wave_id] = idx;
            }
        }
    }
}

// ---- phase 3: global argmin reduction + bit expansion ---------------------

__global__ __launch_bounds__(256) void csi_reduce(const float* __restrict__ candV,
                                                  const int* __restrict__ candI,
                                                  float* __restrict__ out) {
    __shared__ float sv[256];
    __shared__ int   si[256];
    __shared__ int   sidx;
    const int m = blockIdx.x;
    const int tid = threadIdx.x;

    float v = 3.4e38f; int idx = 0x7FFFFFFF;
    for (int wv = tid; wv < NWAVES; wv += 256) {
        float vv = candV[(size_t)m * NWAVES + wv];
        int   ii = candI[(size_t)m * NWAVES + wv];
        lexmin(v, idx, vv, ii);
    }
    sv[tid] = v; si[tid] = idx;
    __syncthreads();
    for (int off = 128; off > 0; off >>= 1) {
        if (tid < off) {
            if (sv[tid + off] < sv[tid] ||
                (sv[tid + off] == sv[tid] && si[tid + off] < si[tid])) {
                sv[tid] = sv[tid + off]; si[tid] = si[tid + off];
            }
        }
        __syncthreads();
    }
    if (tid == 0) sidx = si[0];
    __syncthreads();

    int a, c;
    row_map(m, a, c);
    if (tid < 22)
        out[(size_t)a * 512 + 6 + c * 22 + tid] = (float)((sidx >> tid) & 1);
    if (c == 0 && tid >= 22 && tid < 28)
        out[(size_t)a * 512 + (tid - 22)] = 0.0f;   // leading 6 zeros
}

// ---------------------------------------------------------------------------

extern "C" void kernel_launch(void* const* d_in, const int* in_sizes, int n_in,
                              void* d_out, int out_size, void* d_ws, size_t ws_size,
                              hipStream_t stream) {
    const float* x        = (const float*)d_in[0];   // [16,2,126,128] fp32
    const float* codebook = (const float*)d_in[1];   // [32768,2,48,8] fp32
    float* out = (float*)d_out;                      // [16,512] fp32

    unsigned char* ws     = (unsigned char*)d_ws;
    unsigned char* apack  = ws + WS_APACK;
    float*         rowsum = (float*)(ws + WS_ROWSUM);
    float*         candV  = (float*)(ws + WS_CANDV);
    int*           candI  = (int*)(ws + WS_CANDI);

    csi_pack_a<<<KSTEPS * NTILES, 32, 0, stream>>>(x, apack);
    csi_rowsum<<<2, 256, 0, stream>>>(x, rowsum);
    csi_gemm<<<512, 128, 0, stream>>>(codebook, apack, rowsum, candV, candI);
    csi_reduce<<<NROWS, 256, 0, stream>>>(candV, candI, out);
}
